// AttentionLayer_41901700940426
// MI455X (gfx1250) — matrix operand order repaired and checked by reading, not compiled, not run on hardware
//
#include <hip/hip_runtime.h>
#include <math.h>

typedef __attribute__((ext_vector_type(16))) __bf16 bf16x16;
typedef __attribute__((ext_vector_type(8)))  __bf16 bf16x8;
typedef __attribute__((ext_vector_type(4)))  __bf16 bf16x4;
typedef __attribute__((ext_vector_type(8)))  float  f32x8;

#define HD 512
#define SD 2048
#define BD 64
#define NROWS (SD*BD)     /* 131072 rows, row index = s*BD + b */
#define MROWS 64          /* rows per block */
#define LDA   520         /* padded LDS row stride (bf16): 1040 B = 260 dwords ->
                             rows advance 4 banks; 16 lanes x b128 hit all 64 banks */

// Hardware tanh on CDNA5 (V_TANH_F32 is a TRANS op); cheap fallback otherwise.
__device__ __forceinline__ float fast_tanhf(float x) {
#if __has_builtin(__builtin_amdgcn_tanhf)
    return __builtin_amdgcn_tanhf(x);
#else
    float ax = fabsf(x);
    float e  = __expf(-2.0f * ax);
    float t  = (1.0f - e) * __builtin_amdgcn_rcpf(1.0f + e);
    return copysignf(t, x);
#endif
}

// ---------------------------------------------------------------------------
// Kernel 0a: W (H,H) f32 row-major -> Wt bf16 COLUMN-major: Wt[n*H + k]
// ---------------------------------------------------------------------------
__global__ void k_prep_w(const float* __restrict__ W, __bf16* __restrict__ Wt) {
    int idx = blockIdx.x * blockDim.x + threadIdx.x;   // 0 .. H*H-1
    int n = idx >> 9;
    int k = idx & (HD - 1);
    Wt[idx] = (__bf16)W[k * HD + n];
}

// ---------------------------------------------------------------------------
// Kernel 0b: WR = mean_sent_batch @ W_h  (64x512, tiny) + zero-init out
// ---------------------------------------------------------------------------
__global__ void k_wr(const float* __restrict__ mean, const float* __restrict__ Wh,
                     float* __restrict__ WR, float* __restrict__ out) {
    int b = blockIdx.x, n = threadIdx.x;
    float acc = 0.f;
    for (int k = 0; k < HD; ++k)
        acc = fmaf(mean[b * HD + k], Wh[k * HD + n], acc);
    WR[b * HD + n]  = acc;
    out[b * HD + n] = 0.f;      // out_size == B*H; needed for k_wsum atomics
}

// ---------------------------------------------------------------------------
// Kernel 1: fused  scores[row] = sum_n tanh( (A@W)[row,n] + WR[b,n] ) * ctx[n]
// Block: 64 rows x N(512) x K(512). 8 waves; wave w owns N-tiles [4w,4w+4);
// 4 M-subtiles per wave -> 16 f32x8 accumulators (128 VGPRs).
// A tile staged in LDS as masked bf16 with conflict-free padding (~65 KB).
// W traffic from L2: 2048 blocks x 512 KB = ~1 GB (halved vs 32-row tiles).
// ---------------------------------------------------------------------------
__global__ __launch_bounds__(256) void k_scores(
        const float* __restrict__ sent, const float* __restrict__ mask,
        const __bf16* __restrict__ Wt,  const float* __restrict__ WR,
        const float* __restrict__ context, float* __restrict__ scores) {
    __shared__ __bf16 lA[MROWS * LDA];   // ~65 KB
    __shared__ float  lP[8 * MROWS];     // per-wave partial scores (2 KB)

    const int row0 = blockIdx.x * MROWS;
    const int tid  = threadIdx.x;

    // ---- stage masked A tile as bf16 (float4 granularity) ----
    for (int i = tid; i < MROWS * (HD / 4); i += 256) {
        int rr = i >> 7;            // row in tile
        int c4 = i & 127;           // float4 chunk
        int gr = row0 + rr;
        float m = mask[gr];
        float4 v = ((const float4*)(sent + (size_t)gr * HD))[c4];
        bf16x4 o;
        o[0] = (__bf16)(v.x * m); o[1] = (__bf16)(v.y * m);
        o[2] = (__bf16)(v.z * m); o[3] = (__bf16)(v.w * m);
        ((bf16x4*)&lA[rr * LDA])[c4] = o;
    }
    __syncthreads();

    const int lane = tid & 31;
    const int wv   = tid >> 5;
    const int lr   = lane & 15;   // N within tile / M within subtile
    const int hi   = lane >> 4;   // half-wave select

    f32x8 C[4][4];
    const f32x8 z = {0.f,0.f,0.f,0.f,0.f,0.f,0.f,0.f};
    #pragma unroll
    for (int a = 0; a < 4; ++a)
        #pragma unroll
        for (int t = 0; t < 4; ++t) C[a][t] = z;

    #pragma unroll 2
    for (int kt = 0; kt < HD / 32; ++kt) {
        // A operand (16-bit 16x32): lanes0-15 K0-7|K16-23, lanes16-31 K8-15|K24-31
        bf16x16 a[4];
        {
            int kb = kt * 32 + hi * 8;
            #pragma unroll
            for (int st = 0; st < 4; ++st) {
                const __bf16* p = &lA[(st * 16 + lr) * LDA + kb];
                bf16x8 lo = *(const bf16x8*)p, hh = *(const bf16x8*)(p + 16);
                a[st] = __builtin_shufflevector(lo, hh,
                            0,1,2,3,4,5,6,7,8,9,10,11,12,13,14,15);
            }
        }
        #pragma unroll
        for (int t = 0; t < 4; ++t) {
            // B operand (16-bit 32x16): lanes0-15 K0-7|K16-23, lanes16-31 K8-15|K24-31
            int n  = (wv * 4 + t) * 16 + lr;
            int kb = kt * 32 + hi * 8;
            const __bf16* q = &Wt[n * HD + kb];
            bf16x8 b0 = *(const bf16x8*)q, b1 = *(const bf16x8*)(q + 16);
            bf16x16 bm = __builtin_shufflevector(b0, b1,
                            0,1,2,3,4,5,6,7,8,9,10,11,12,13,14,15);
            #pragma unroll
            for (int st = 0; st < 4; ++st)
                C[st][t] = __builtin_amdgcn_wmma_f32_16x16x32_bf16(
                               false, a[st], false, bm, (short)0, C[st][t],
                               false, false);
        }
    }

    // ---- epilogue: + WR gather, hw tanh, dot with context ----
    float sacc[4][8];
    #pragma unroll
    for (int st = 0; st < 4; ++st)
        #pragma unroll
        for (int v = 0; v < 8; ++v) sacc[st][v] = 0.f;

    #pragma unroll
    for (int t = 0; t < 4; ++t) {
        int n = (wv * 4 + t) * 16 + lr;
        float ctx = context[n];
        #pragma unroll
        for (int st = 0; st < 4; ++st) {
            #pragma unroll
            for (int v = 0; v < 8; ++v) {
                int rl = st * 16 + v + 8 * hi;             // local row = b (row0 is 64-aligned)
                float x = C[st][t][v] + WR[rl * HD + n];
                sacc[st][v] += fast_tanhf(x) * ctx;
            }
        }
    }
    // reduce over the 16 N-lanes of each half-wave
    #pragma unroll
    for (int st = 0; st < 4; ++st)
        #pragma unroll
        for (int v = 0; v < 8; ++v) {
            float s = sacc[st][v];
            s += __shfl_xor(s, 1, 32);
            s += __shfl_xor(s, 2, 32);
            s += __shfl_xor(s, 4, 32);
            s += __shfl_xor(s, 8, 32);
            sacc[st][v] = s;
        }
    if (lr == 0) {   // lanes 0 and 16 hold the half-wave sums
        #pragma unroll
        for (int st = 0; st < 4; ++st)
            #pragma unroll
            for (int v = 0; v < 8; ++v)
                lP[wv * MROWS + st * 16 + 8 * hi + v] = sacc[st][v];
    }
    __syncthreads();
    if (tid < MROWS) {  // cross-wave reduction (waves hold disjoint N ranges)
        float s = 0.f;
        #pragma unroll
        for (int w2 = 0; w2 < 8; ++w2) s += lP[w2 * MROWS + tid];
        scores[row0 + tid] = s;
    }
}

// ---------------------------------------------------------------------------
// Kernel 2: softmax over S (stride BD) for each b, in place.
// ---------------------------------------------------------------------------
__global__ void k_softmax(float* __restrict__ sc) {
    const int b = blockIdx.x, t = threadIdx.x;   // 256 threads, 8 elems each
    __shared__ float red[256];
    float v[8];
    float mx = -3.4e38f;
    #pragma unroll
    for (int i = 0; i < 8; ++i) {
        v[i] = sc[(t + i * 256) * BD + b];
        mx = fmaxf(mx, v[i]);
    }
    red[t] = mx; __syncthreads();
    for (int s2 = 128; s2 > 0; s2 >>= 1) {
        if (t < s2) red[t] = fmaxf(red[t], red[t + s2]);
        __syncthreads();
    }
    mx = red[0]; __syncthreads();
    float sum = 0.f;
    #pragma unroll
    for (int i = 0; i < 8; ++i) { v[i] = __expf(v[i] - mx); sum += v[i]; }
    red[t] = sum; __syncthreads();
    for (int s2 = 128; s2 > 0; s2 >>= 1) {
        if (t < s2) red[t] += red[t + s2];
        __syncthreads();
    }
    float inv = __builtin_amdgcn_rcpf(red[0]);
    #pragma unroll
    for (int i = 0; i < 8; ++i) sc[(t + i * 256) * BD + b] = v[i] * inv;
}

// ---------------------------------------------------------------------------
// Kernel 3: out[b,h] += sum_s alpha[s,b]*mask[s,b]*sent[s,b,h]  (bandwidth)
// ---------------------------------------------------------------------------
__global__ void k_wsum(const float* __restrict__ sent, const float* __restrict__ mask,
                       const float* __restrict__ alpha, float* __restrict__ out) {
    const int h  = threadIdx.x;
    const int b  = blockIdx.y;
    const int s0 = blockIdx.x * (SD / 16);
    float acc = 0.f;
    for (int s = s0; s < s0 + SD / 16; ++s) {
        int rb = s * BD + b;
        float am = alpha[rb] * mask[rb];
        acc = fmaf(am, sent[(size_t)rb * HD + h], acc);
    }
    atomicAdd(&out[b * HD + h], acc);
}

// ---------------------------------------------------------------------------
extern "C" void kernel_launch(void* const* d_in, const int* in_sizes, int n_in,
                              void* d_out, int out_size, void* d_ws, size_t ws_size,
                              hipStream_t stream) {
    const float* sent = (const float*)d_in[0];   // (S,B,H) f32
    const float* mean = (const float*)d_in[1];   // (B,H)
    const float* mask = (const float*)d_in[2];   // (S,B)
    const float* W    = (const float*)d_in[3];   // (H,H)
    const float* Wh   = (const float*)d_in[4];   // (H,H)
    const float* ctx  = (const float*)d_in[5];   // (H,)
    float* out = (float*)d_out;                  // (B,H)

    char* ws = (char*)d_ws;
    __bf16* Wt = (__bf16*)ws;                            // 512 KB bf16 W, col-major
    float*  WR = (float*)(ws + 512 * 1024);              // 128 KB
    float*  sc = (float*)(ws + 512 * 1024 + 128 * 1024); // 512 KB scores/alpha

    k_prep_w <<<dim3(HD * HD / 256), dim3(256), 0, stream>>>(W, Wt);
    k_wr     <<<dim3(BD),            dim3(HD),  0, stream>>>(mean, Wh, WR, out);
    k_scores <<<dim3(NROWS / MROWS), dim3(256), 0, stream>>>(sent, mask, Wt, WR, ctx, sc);
    k_softmax<<<dim3(BD),            dim3(256), 0, stream>>>(sc);
    k_wsum   <<<dim3(16, BD),        dim3(HD),  0, stream>>>(sent, mask, sc, out);
}
